// HypAgg_39410619908630
// MI455X (gfx1250) — matrix operand order repaired and checked by reading, not compiled, not run on hardware
//
#include <hip/hip_runtime.h>

// Problem constants (match reference setup_inputs)
#define NB   2
#define NN   8192
#define ND   256
#define BM   64          // rows per workgroup tile
#define BK   32          // K per WMMA step (bf16 16x16x32)
#define LDA  40          // LDS A row stride in bf16 elems (80 B, padded vs 64 B)

typedef __bf16 v16bf __attribute__((ext_vector_type(16)));
typedef float  v8f   __attribute__((ext_vector_type(8)));

// f32 -> bf16 (round-to-nearest, half-away ties) -- 1 v_add per element plus
// a single v_perm_b32 to pack the two high halves: 3 VALU per packed pair.
__device__ __forceinline__ unsigned pk_bf16x2(float lo, float hi) {
  unsigned a = __builtin_bit_cast(unsigned, lo) + 0x8000u;
  unsigned b = __builtin_bit_cast(unsigned, hi) + 0x8000u;
  // result = { b[31:16], a[31:16] }  (src0 = b -> sel bytes 6,7; src1 = a -> 2,3)
  return __builtin_amdgcn_perm(b, a, 0x07060302u);
}
__device__ __forceinline__ unsigned short f32_bf16(float f) {
  return (unsigned short)((__builtin_bit_cast(unsigned, f) + 0x8000u) >> 16);
}

// ---------------------------------------------------------------------------
// Kernel 1: x_tangent = logmap0(x)  (c = 1)
// Writes bf16 TRANSPOSED layout xtT[b][d][k] so the GEMM's WMMA B-fragments
// are plain contiguous 32B loads in the same contiguous k order as the A fragments.
// One wave (32 lanes) per row; lane handles 8 of the 256 features.
// ---------------------------------------------------------------------------
__global__ __launch_bounds__(256) void logmap0_t_kernel(
    const float* __restrict__ x, unsigned short* __restrict__ xtT) {
  const int lane = threadIdx.x & 31;
  const int row  = blockIdx.x * 8 + (threadIdx.x >> 5);   // 0 .. NB*NN-1
  const int b = row >> 13;                                 // NN = 8192 = 2^13
  const int n = row & (NN - 1);

  const float* xr = x + (size_t)row * ND + lane * 8;
  float4 u0 = *(const float4*)xr;
  float4 u1 = *(const float4*)(xr + 4);

  float ss = u0.x*u0.x + u0.y*u0.y + u0.z*u0.z + u0.w*u0.w
           + u1.x*u1.x + u1.y*u1.y + u1.z*u1.z + u1.w*u1.w;
#pragma unroll
  for (int m = 16; m >= 1; m >>= 1) ss += __shfl_xor(ss, m, 32);

  float nn  = sqrtf(ss);
  float ncl = fmaxf(nn, 1e-15f);                 // _norm clip
  float z   = fminf(ncl, 1.0f - 1e-7f);          // artanh clip (n >= 0)
  float s   = atanhf(z) / ncl;                   // artanh(||x||)/||x||

  float v[8] = {u0.x, u0.y, u0.z, u0.w, u1.x, u1.y, u1.z, u1.w};
  unsigned short* dst = xtT + (size_t)b * ND * NN + (size_t)(lane * 8) * NN + n;
#pragma unroll
  for (int j = 0; j < 8; ++j) dst[(size_t)j * NN] = f32_bf16(v[j] * s);
}

// ---------------------------------------------------------------------------
// Kernel 2: out = project(expmap0(adj @ xt)), WMMA bf16 -> f32 accumulate.
// Grid: (NN/BM, NB). 256 threads = 8 waves.
// Wave w owns M-tiles {2*(w&1), 2*(w&1)+1} and N-cols [(w>>1)*64, +64).
// adj tile (64x32 f32) -> bf16 in padded LDS; xtT fragments direct from global.
// Fused epilogue: per-row ||.||^2 via shuffles + ds_add_f32, tanh + ball clip.
// ---------------------------------------------------------------------------
__global__ __launch_bounds__(256) void hypagg_wmma_kernel(
    const float* __restrict__ adj, const unsigned short* __restrict__ xtT,
    float* __restrict__ out) {
  __shared__ unsigned short As[BM * LDA];   // 5120 B, padded rows (80 B)
  __shared__ float rowss[BM];               // per-row sum of squares

  const int tid  = threadIdx.x;
  const int lane = tid & 31;
  const int w    = tid >> 5;
  const int b    = blockIdx.y;
  const int rowBlock = blockIdx.x * BM;

  const float*          adjb = adj + (size_t)b * NN * NN;
  const unsigned short* xb   = xtT + (size_t)b * ND * NN;

  const int mtA = (w & 1) * 2;        // first of two 16-row tiles for this wave
  const int d0  = (w >> 1) * 64;      // first of four 16-col tiles

  if (tid < BM) rowss[tid] = 0.0f;

  v8f acc[2][4];
  const v8f vzero = {0,0,0,0,0,0,0,0};
#pragma unroll
  for (int mi = 0; mi < 2; ++mi)
#pragma unroll
    for (int j = 0; j < 4; ++j) acc[mi][j] = vzero;

  const int ar = tid >> 2;            // cooperative adj-tile row   (0..63)
  const int ac = (tid & 3) * 8;       // cooperative adj-tile col   {0,8,16,24}

  for (int k0 = 0; k0 < NN; k0 += BK) {
    __syncthreads();
    {
      // load 64x32 f32 adj tile, convert to bf16 (add + v_perm), stage in LDS
      const float* src = adjb + (size_t)(rowBlock + ar) * NN + (k0 + ac);
      float4 f0 = *(const float4*)src;
      float4 f1 = *(const float4*)(src + 4);
      uint4 p;
      p.x = pk_bf16x2(f0.x, f0.y);
      p.y = pk_bf16x2(f0.z, f0.w);
      p.z = pk_bf16x2(f1.x, f1.y);
      p.w = pk_bf16x2(f1.z, f1.w);
      *(uint4*)((char*)As + ar * (LDA * 2) + ac * 2) = p;
      if (k0 + BK < NN) __builtin_prefetch(src + BK, 0, 1);  // next k-panel
    }
    __syncthreads();

    // B fragments: lane = column (l&15), K = 16*(l>>4) + 0..15, contiguous
    // 32B in transposed xtT -> one aligned v16bf load each.
    v16bf bf[4];
#pragma unroll
    for (int j = 0; j < 4; ++j) {
      const unsigned short* bp =
          xb + (size_t)(d0 + j * 16 + (lane & 15)) * NN + k0 + ((lane >> 4) << 4);
      bf[j] = *(const v16bf*)bp;
    }

    // A fragments from LDS: lane l holds row M=(l&15), K {base..base+15}
    // with base = 16*(l>>4)  -> two ds_load_b128.
    union { uint4 u[2]; v16bf v; } af[2];
#pragma unroll
    for (int mi = 0; mi < 2; ++mi) {
      const int mrow = (mtA + mi) * 16 + (lane & 15);
      const char* ap = (const char*)As + mrow * (LDA * 2) + ((lane >> 4) << 5);
      af[mi].u[0] = *(const uint4*)ap;
      af[mi].u[1] = *(const uint4*)(ap + 16);
    }

#pragma unroll
    for (int mi = 0; mi < 2; ++mi)
#pragma unroll
      for (int j = 0; j < 4; ++j)
        acc[mi][j] = __builtin_amdgcn_wmma_f32_16x16x32_bf16(
            false, af[mi].v, false, bf[j], (short)0, acc[mi][j], false, false);
  }

  __syncthreads();

  // Partial sum-of-squares: for fixed (mi,v), lanes 0-15 all hold row
  // (mtA+mi)*16+v and lanes 16-31 row +8; reduce each half, one lane adds.
#pragma unroll
  for (int mi = 0; mi < 2; ++mi)
#pragma unroll
    for (int v = 0; v < 8; ++v) {
      float p = 0.0f;
#pragma unroll
      for (int j = 0; j < 4; ++j) { float t = acc[mi][j][v]; p += t * t; }
#pragma unroll
      for (int m = 1; m < 16; m <<= 1) p += __shfl_xor(p, m, 32);
      if ((lane & 15) == 0)
        atomicAdd(&rowss[(mtA + mi) * 16 + (lane >> 4) * 8 + v], p);
    }
  __syncthreads();

  // expmap0 + project: scale = min(tanh(||u||), 1-4e-3) / max(||u||,1e-15)
#pragma unroll
  for (int mi = 0; mi < 2; ++mi)
#pragma unroll
    for (int v = 0; v < 8; ++v) {
      const int r  = (mtA + mi) * 16 + (lane >> 4) * 8 + v;
      float ncl = fmaxf(sqrtf(rowss[r]), 1e-15f);
      float s   = fminf(tanhf(ncl), 1.0f - 4e-3f) / ncl;
      float* o = out + ((size_t)b * NN + rowBlock + r) * ND + d0 + (lane & 15);
#pragma unroll
      for (int j = 0; j < 4; ++j) o[j * 16] = acc[mi][j][v] * s;
    }
}

// ---------------------------------------------------------------------------
extern "C" void kernel_launch(void* const* d_in, const int* in_sizes, int n_in,
                              void* d_out, int out_size, void* d_ws, size_t ws_size,
                              hipStream_t stream) {
  const float* x   = (const float*)d_in[0];   // [2, 8192, 256] f32
  const float* adj = (const float*)d_in[1];   // [2, 8192, 8192] f32
  float* out = (float*)d_out;                 // [2, 8192, 256] f32
  unsigned short* xtT = (unsigned short*)d_ws; // bf16 [2][256][8192] = 8 MB

  logmap0_t_kernel<<<(NB * NN) / 8, 256, 0, stream>>>(x, xtT);
  hypagg_wmma_kernel<<<dim3(NN / BM, NB), 256, 0, stream>>>(adj, xtT, out);
}
